// SpatialAttn2_6674379178280
// MI455X (gfx1250) — matrix operand order repaired and checked by reading, not compiled, not run on hardware
//
#include <hip/hip_runtime.h>
#include <hip/hip_bf16.h>
#include <math.h>

typedef __attribute__((ext_vector_type(16))) _Float16 v16h;
typedef __attribute__((ext_vector_type(4)))  _Float16 v4h;
typedef __attribute__((ext_vector_type(8)))  float    v8f;
typedef __attribute__((ext_vector_type(2)))  float    v2f;

#ifndef __has_builtin
#define __has_builtin(x) 0
#endif
#if __has_builtin(__builtin_amdgcn_wmma_f32_16x16x4_f32)
#define HAS_WMMA_F32X4 1
#else
#define HAS_WMMA_F32X4 0
#endif

#define BATCH   8
#define LSEQ    24
#define NSEQ    512
#define PAST    12
#define FUTURE  4
#define HEAD    4
#define HIDDEN  32
#define STEPS   (LSEQ - PAST)        // 12
#define BS      (BATCH * STEPS)      // 96
#define NEGV    (-1.0e9f)

// -------------------------------------------------------------------------
// Kernel 1: layer-1 Q/K/V projections.  x[bs,n,p] = inp[b, s+p, n]
// q1 is pre-scaled by 1/sqrt(32) so the WMMA score needs no epilogue mul.
// q1/k1 stored [bs][h][n][32] f16, v stored transposed vT1[bs][h][o][n] f16.
// -------------------------------------------------------------------------
__global__ __launch_bounds__(256)
void qkv1_kernel(const float* __restrict__ inp,
                 const float* __restrict__ Wq, const float* __restrict__ Wk,
                 const float* __restrict__ Wv,
                 _Float16* __restrict__ q1, _Float16* __restrict__ k1,
                 _Float16* __restrict__ vT1)
{
    __shared__ float xs[PAST * NSEQ];              // 24 KB
    const int wg = blockIdx.x;                     // bs*4 + h
    const int bs = wg >> 2;
    const int h  = wg & 3;
    const int b  = bs / STEPS, s = bs % STEPS;
    const int tid = threadIdx.x;
    const float scale = 0.1767766952966369f;       // 1/sqrt(32)

    for (int i = tid; i < PAST * NSEQ; i += 256)
        xs[i] = inp[((size_t)b * LSEQ + s + (i >> 9)) * NSEQ + (i & 511)];
    __syncthreads();

    const int o  = tid & 31;
    const int tn = tid >> 5;                       // 0..7
    float wq[PAST], wk[PAST], wv[PAST];
#pragma unroll
    for (int p = 0; p < PAST; ++p) {
        wq[p] = Wq[(h * PAST + p) * HIDDEN + o] * scale;   // fold 1/sqrt(d)
        wk[p] = Wk[(h * PAST + p) * HIDDEN + o];
        wv[p] = Wv[(h * PAST + p) * HIDDEN + o];
    }
    const size_t bh = (size_t)bs * HEAD + h;
    for (int nb = 0; nb < NSEQ; nb += 8) {
        const int n = nb + tn;
        float aq = 0.f, ak = 0.f, av = 0.f;
#pragma unroll
        for (int p = 0; p < PAST; ++p) {
            const float xv = xs[p * NSEQ + n];
            aq = fmaf(xv, wq[p], aq);
            ak = fmaf(xv, wk[p], ak);
            av = fmaf(xv, wv[p], av);
        }
        q1[(bh * NSEQ + n) * HIDDEN + o]  = (_Float16)aq;
        k1[(bh * NSEQ + n) * HIDDEN + o]  = (_Float16)ak;
        vT1[(bh * HIDDEN + o) * NSEQ + n] = (_Float16)av;
    }
}

// -------------------------------------------------------------------------
// Kernel 2: layer-1 attention.  2 waves per block, one (bs, h, 16-row slab).
// Wave w handles score tiles mt in [w*16, w*16+16) and ctx N-tile t==w.
// Scores via v_wmma_f32_16x16x32_f16 (scale pre-folded into q1);
// mask/weight applied in a coalesced int4/float4 pass; softmax via LDS
// partials; ctx via 16 more f16 WMMAs per wave.
// -------------------------------------------------------------------------
__global__ __launch_bounds__(64)
void attn1_kernel(const _Float16* __restrict__ q1, const _Float16* __restrict__ k1,
                  const _Float16* __restrict__ vT1, const int* __restrict__ mask,
                  float* __restrict__ attn1, float* __restrict__ weight1,
                  float* __restrict__ o1)
{
    __shared__ float    sc[16][NSEQ];      // 32 KB scores
    __shared__ _Float16 att[16][NSEQ];     // 16 KB f16 probabilities
    __shared__ float    pm[4][16], ps[4][16];
    __shared__ float    rmax[16], rinv[16];

    const int wg   = blockIdx.x;           // bs*128 + h*32 + nblk
    const int bs   = wg >> 7;
    const int h    = (wg >> 5) & 3;
    const int nblk = wg & 31;
    const int n0   = nblk * 16;
    const int tid  = threadIdx.x;          // 0..63
    const int w    = tid >> 5;             // wave id 0/1
    const int lane = tid & 31;
    const int lrow = lane & 15;
    const int hi   = lane >> 4;            // half of wave
    const int b    = bs / STEPS, s = bs % STEPS;
    const bool last = (s == STEPS - 1);
    const size_t bh = (size_t)bs * HEAD + h;

    // ---- A operand: q rows n0..n0+15, K contiguous 16 per half (k = 16*hi + j, same order as B)
    v16h a;
    {
        const _Float16* qr = q1 + (bh * NSEQ + n0 + lrow) * HIDDEN;
        const int c0 = hi ? 16 : 0;
#pragma unroll
        for (int j = 0; j < 16; ++j) a[j] = qr[c0 + j];
    }

    // ---- scores: 16 column tiles per wave, one f16 WMMA each (raw scores)
    const int mtBeg = w * 16, mtEnd = mtBeg + 16;
    for (int mt = mtBeg; mt < mtEnd; ++mt) {
        const int m0 = mt * 16;
        const _Float16* kr = k1 + (bh * NSEQ + m0 + lrow) * HIDDEN;
        v16h bm;
        {
            const int cb = hi ? 16 : 0;
#pragma unroll
            for (int j = 0; j < 16; ++j) bm[j] = kr[cb + j];
        }
        // speculative prefetch of next K tile row; past-end is silently dropped
        __builtin_prefetch(kr + 16 * HIDDEN, 0, 1);
        v8f acc = {};
        acc = __builtin_amdgcn_wmma_f32_16x16x32_f16(false, a, false, bm,
                                                     (short)0, acc, false, false);
        const int mo = hi * 8;
        const int gm = m0 + lrow;
#pragma unroll
        for (int r = 0; r < 8; ++r) sc[r + mo][gm] = acc[r];
    }
    __syncthreads();

    // ---- coalesced mask (+ weight1) pass, 16B wide
    {
        const int* mbase = mask + n0 * NSEQ;
        if (last) {
            float* wbase = weight1 + (((size_t)b * HEAD + h) * NSEQ + n0) * NSEQ;
            for (int idx = tid; idx < 16 * 128; idx += 64) {
                const int rr = idx >> 7, m = (idx & 127) << 2;
                const int lin = (rr << 9) + m;
                const int4  mm = *(const int4*)(mbase + lin);
                float4 v = *(const float4*)&sc[rr][m];
                v.x = mm.x > 0 ? v.x : NEGV;
                v.y = mm.y > 0 ? v.y : NEGV;
                v.z = mm.z > 0 ? v.z : NEGV;
                v.w = mm.w > 0 ? v.w : NEGV;
                *(float4*)&sc[rr][m] = v;
                *(float4*)(wbase + lin) = v;
            }
        } else {
            for (int idx = tid; idx < 16 * 128; idx += 64) {
                const int rr = idx >> 7, m = (idx & 127) << 2;
                const int lin = (rr << 9) + m;
                const int4  mm = *(const int4*)(mbase + lin);
                float4 v = *(const float4*)&sc[rr][m];
                v.x = mm.x > 0 ? v.x : NEGV;
                v.y = mm.y > 0 ? v.y : NEGV;
                v.z = mm.z > 0 ? v.z : NEGV;
                v.w = mm.w > 0 ? v.w : NEGV;
                *(float4*)&sc[rr][m] = v;
            }
        }
    }
    __syncthreads();

    // ---- softmax: 4 lanes per row (quarter q = tid>>4), LDS partial reduce
    {
        const int q  = tid >> 4;
        const int rr = tid & 15;
        const int ms = q * 128;
        float mx = -3.0e38f;
        for (int m4 = 0; m4 < 32; ++m4) {
            const float4 t = *(const float4*)&sc[rr][ms + m4 * 4];
            mx = fmaxf(mx, fmaxf(fmaxf(t.x, t.y), fmaxf(t.z, t.w)));
        }
        pm[q][rr] = mx;
        __syncthreads();
        if (tid < 16)
            rmax[tid] = fmaxf(fmaxf(pm[0][tid], pm[1][tid]),
                              fmaxf(pm[2][tid], pm[3][tid]));
        __syncthreads();
        const float rm = rmax[rr];
        float sum = 0.f;
        for (int m4 = 0; m4 < 32; ++m4) {
            const float4 t = *(const float4*)&sc[rr][ms + m4 * 4];
            sum += __expf(t.x - rm) + __expf(t.y - rm)
                 + __expf(t.z - rm) + __expf(t.w - rm);
        }
        ps[q][rr] = sum;
        __syncthreads();
        if (tid < 16)
            rinv[tid] = 1.0f / (ps[0][tid] + ps[1][tid] + ps[2][tid] + ps[3][tid]);
        __syncthreads();
    }

    // ---- coalesced normalize: b128 attn1 store + packed f16 LDS copy
    {
        float* abase = attn1 + (bh * NSEQ + n0) * NSEQ;
        for (int idx = tid; idx < 16 * 128; idx += 64) {
            const int rr = idx >> 7, m = (idx & 127) << 2;
            const float4 t = *(const float4*)&sc[rr][m];
            const float rm = rmax[rr], ri = rinv[rr];
            float4 p;
            p.x = __expf(t.x - rm) * ri;
            p.y = __expf(t.y - rm) * ri;
            p.z = __expf(t.z - rm) * ri;
            p.w = __expf(t.w - rm) * ri;
            *(float4*)(abase + (rr << 9) + m) = p;
            v4h ph;
            ph[0] = (_Float16)p.x; ph[1] = (_Float16)p.y;
            ph[2] = (_Float16)p.z; ph[3] = (_Float16)p.w;
            *(v4h*)&att[rr][m] = ph;
        }
    }
    __syncthreads();

    // ---- ctx = P(16x512) * V(512x32): wave w owns N-tile t==w (cols w*16..)
    {
        v8f acc2 = {};
        const _Float16* vr = vT1 + (bh * HIDDEN + w * 16 + lrow) * NSEQ;
        for (int kt = 0; kt < 16; ++kt) {
            v16h aa;
            const int c0 = kt * 32 + (hi ? 16 : 0);
#pragma unroll
            for (int j = 0; j < 16; ++j) {
                aa[j] = att[lrow][c0 + j];
            }
            v16h bb;
            const int cb = kt * 32 + (hi ? 16 : 0);
#pragma unroll
            for (int j = 0; j < 16; ++j) bb[j] = vr[cb + j];
            acc2 = __builtin_amdgcn_wmma_f32_16x16x32_f16(false, aa, false, bb,
                                                          (short)0, acc2, false, false);
        }
        const int mo = hi * 8;
#pragma unroll
        for (int r = 0; r < 8; ++r)
            o1[((size_t)bs * NSEQ + n0 + mo + r) * (HEAD * HIDDEN)
               + h * HIDDEN + w * 16 + lrow] = acc2[r];
    }
}

// -------------------------------------------------------------------------
// Kernel 3: layer-2 projections (128 -> 4), plain VALU (tiny).
// q2 pre-scaled by 1/sqrt(4) = 0.5 (applied at the q2 store).
// -------------------------------------------------------------------------
__global__ __launch_bounds__(256)
void qkv2_kernel(const float* __restrict__ o1,
                 const float* __restrict__ Wq2, const float* __restrict__ Wk2,
                 const float* __restrict__ Wv2,
                 float* __restrict__ q2, float* __restrict__ k2, float* __restrict__ v2)
{
    const int i = blockIdx.x * 256 + threadIdx.x;    // (bs, n) flat
    if (i >= BS * NSEQ) return;
    const float* xr = o1 + (size_t)i * (HEAD * HIDDEN);
    float aq[4] = {}, ak[4] = {}, av[4] = {};
    for (int kk = 0; kk < HEAD * HIDDEN; ++kk) {
        const float xv = xr[kk];
#pragma unroll
        for (int f = 0; f < 4; ++f) {
            aq[f] = fmaf(xv, Wq2[kk * 4 + f], aq[f]);
            ak[f] = fmaf(xv, Wk2[kk * 4 + f], ak[f]);
            av[f] = fmaf(xv, Wv2[kk * 4 + f], av[f]);
        }
    }
#pragma unroll
    for (int f = 0; f < 4; ++f) {
        q2[(size_t)i * 4 + f] = aq[f] * 0.5f;        // fold 1/sqrt(4)
        k2[(size_t)i * 4 + f] = ak[f];
        v2[(size_t)i * 4 + f] = av[f];
    }
}

// -------------------------------------------------------------------------
// Kernel 4: layer-2 attention.  v_wmma_f32_16x16x4_f32 (K=4 exact, f32),
// coalesced mask/weight pass, softmax, VALU ctx, transposed output write.
// -------------------------------------------------------------------------
__global__ __launch_bounds__(32)
void attn2_kernel(const float* __restrict__ q2, const float* __restrict__ k2,
                  const float* __restrict__ v2, const int* __restrict__ mask,
                  float* __restrict__ attn2, float* __restrict__ weight2,
                  float* __restrict__ outp)
{
    __shared__ float sc[16][NSEQ];         // 32 KB (scores, then probs)
    __shared__ float rmax[16], rinv[16];

    const int wg   = blockIdx.x;           // bs*32 + nblk
    const int bs   = wg >> 5;
    const int nblk = wg & 31;
    const int n0   = nblk * 16;
    const int lane = threadIdx.x;
    const int lrow = lane & 15;
    const int hi   = lane >> 4;
    const int b    = bs / STEPS, s = bs % STEPS;
    const bool last = (s == STEPS - 1);
    const float* qb = q2 + (size_t)bs * NSEQ * 4;
    const float* kb = k2 + (size_t)bs * NSEQ * 4;

#if HAS_WMMA_F32X4
    v2f a2;
    {   // A: 16x4 f32 -> lanes 0-15 K={0,1}, lanes 16-31 K={2,3}
        const float* qr = qb + (n0 + lrow) * 4 + (hi ? 2 : 0);
        a2[0] = qr[0]; a2[1] = qr[1];
    }
#endif
    for (int mt = 0; mt < 32; ++mt) {
        const int m0 = mt * 16;
        const int gm = m0 + lrow;
        const int mo = hi * 8;
#if HAS_WMMA_F32X4
        v2f b2;
        {
            const float* kr = kb + gm * 4 + (hi ? 2 : 0);
            b2[0] = kr[0]; b2[1] = kr[1];
        }
        v8f acc = {};
        acc = __builtin_amdgcn_wmma_f32_16x16x4_f32(false, a2, false, b2,
                                                    (short)0, acc, false, false);
#pragma unroll
        for (int r = 0; r < 8; ++r) sc[r + mo][gm] = acc[r];
#else
#pragma unroll
        for (int r = 0; r < 8; ++r) {
            const int nl = r + mo;
            float v = 0.f;
#pragma unroll
            for (int kk = 0; kk < 4; ++kk)
                v = fmaf(qb[(n0 + nl) * 4 + kk], kb[gm * 4 + kk], v);
            sc[nl][gm] = v;
        }
#endif
    }
    __syncthreads();

    // ---- coalesced mask (+ weight2), 16B wide
    {
        const int* mbase = mask + n0 * NSEQ;
        if (last) {
            float* wbase = weight2 + ((size_t)b * NSEQ + n0) * NSEQ;
            for (int idx = lane; idx < 16 * 128; idx += 32) {
                const int rr = idx >> 7, m = (idx & 127) << 2;
                const int lin = (rr << 9) + m;
                const int4  mm = *(const int4*)(mbase + lin);
                float4 v = *(const float4*)&sc[rr][m];
                v.x = mm.x > 0 ? v.x : NEGV;
                v.y = mm.y > 0 ? v.y : NEGV;
                v.z = mm.z > 0 ? v.z : NEGV;
                v.w = mm.w > 0 ? v.w : NEGV;
                *(float4*)&sc[rr][m] = v;
                *(float4*)(wbase + lin) = v;
            }
        } else {
            for (int idx = lane; idx < 16 * 128; idx += 32) {
                const int rr = idx >> 7, m = (idx & 127) << 2;
                const int4  mm = *(const int4*)(mbase + (rr << 9) + m);
                float4 v = *(const float4*)&sc[rr][m];
                v.x = mm.x > 0 ? v.x : NEGV;
                v.y = mm.y > 0 ? v.y : NEGV;
                v.z = mm.z > 0 ? v.z : NEGV;
                v.w = mm.w > 0 ? v.w : NEGV;
                *(float4*)&sc[rr][m] = v;
            }
        }
    }
    __syncthreads();

    {   // softmax: 2 lanes per row, combined via wave32 shfl_xor(16)
        const int ms = hi * 256;
        float mx = -3.0e38f;
        for (int m4 = 0; m4 < 64; ++m4) {
            const float4 t = *(const float4*)&sc[lrow][ms + m4 * 4];
            mx = fmaxf(mx, fmaxf(fmaxf(t.x, t.y), fmaxf(t.z, t.w)));
        }
        mx = fmaxf(mx, __shfl_xor(mx, 16, 32));
        float sum = 0.f;
        for (int m4 = 0; m4 < 64; ++m4) {
            const float4 t = *(const float4*)&sc[lrow][ms + m4 * 4];
            sum += __expf(t.x - mx) + __expf(t.y - mx)
                 + __expf(t.z - mx) + __expf(t.w - mx);
        }
        sum += __shfl_xor(sum, 16, 32);
        if (hi == 0) { rmax[lrow] = mx; rinv[lrow] = 1.0f / sum; }
    }
    __syncthreads();

    {   // coalesced normalize: sc becomes probabilities; b128 attn2 store
        float* abase = attn2 + ((size_t)bs * NSEQ + n0) * NSEQ;
        for (int idx = lane; idx < 16 * 128; idx += 32) {
            const int rr = idx >> 7, m = (idx & 127) << 2;
            const float4 t = *(const float4*)&sc[rr][m];
            const float rm = rmax[rr], ri = rinv[rr];
            float4 p;
            p.x = __expf(t.x - rm) * ri;
            p.y = __expf(t.y - rm) * ri;
            p.z = __expf(t.z - rm) * ri;
            p.w = __expf(t.w - rm) * ri;
            *(float4*)&sc[rr][m] = p;
            *(float4*)(abase + (rr << 9) + m) = p;
        }
    }
    __syncthreads();

    {   // ctx2: 16 rows x 4 outputs, 2 per lane; out[b,s,f,n] = ctx2[bs,n,f]
        const float* vb = v2 + (size_t)bs * NSEQ * 4;
        const int f0 = hi * 2;
        float a0 = 0.f, a1 = 0.f;
        for (int m = 0; m < NSEQ; ++m) {
            const float p = sc[lrow][m];
            a0 = fmaf(p, vb[m * 4 + f0], a0);
            a1 = fmaf(p, vb[m * 4 + f0 + 1], a1);
        }
        outp[((size_t)bs * FUTURE + f0) * NSEQ + n0 + lrow]     = a0;
        outp[((size_t)bs * FUTURE + f0 + 1) * NSEQ + n0 + lrow] = a1;
    }
}

// -------------------------------------------------------------------------
extern "C" void kernel_launch(void* const* d_in, const int* in_sizes, int n_in,
                              void* d_out, int out_size, void* d_ws, size_t ws_size,
                              hipStream_t stream)
{
    const float* inp  = (const float*)d_in[0];
    const int*   mask = (const int*)  d_in[1];
    const float* Wq1  = (const float*)d_in[2];
    const float* Wk1  = (const float*)d_in[3];
    const float* Wv1  = (const float*)d_in[4];
    const float* Wq2  = (const float*)d_in[5];
    const float* Wk2  = (const float*)d_in[6];
    const float* Wv2  = (const float*)d_in[7];

    float* out     = (float*)d_out;                       // 8*12*4*512
    float* attn1   = out + (size_t)BATCH * STEPS * FUTURE * NSEQ;
    float* attn2   = attn1 + (size_t)BS * HEAD * NSEQ * NSEQ;
    float* weight1 = attn2 + (size_t)BS * NSEQ * NSEQ;
    float* weight2 = weight1 + (size_t)BATCH * HEAD * NSEQ * NSEQ;

    char* ws = (char*)d_ws;
    const size_t qkvHalfBytes = (size_t)BS * HEAD * NSEQ * HIDDEN * 2;  // 12.58 MB
    _Float16* q1  = (_Float16*)(ws);
    _Float16* k1  = (_Float16*)(ws + qkvHalfBytes);
    _Float16* vT1 = (_Float16*)(ws + 2 * qkvHalfBytes);
    float*    o1  = (float*)   (ws + 3 * qkvHalfBytes);                 // 25.17 MB
    char* p2 = ws + 3 * qkvHalfBytes + (size_t)BS * NSEQ * HEAD * HIDDEN * 4;
    float* q2 = (float*)(p2);
    float* k2 = (float*)(p2 + (size_t)BS * NSEQ * 4 * 4);
    float* v2 = (float*)(p2 + 2 * (size_t)BS * NSEQ * 4 * 4);

    qkv1_kernel <<<BS * HEAD,             256, 0, stream>>>(inp, Wq1, Wk1, Wv1, q1, k1, vT1);
    attn1_kernel<<<BS * HEAD * (NSEQ/16),  64, 0, stream>>>(q1, k1, vT1, mask, attn1, weight1, o1);
    qkv2_kernel <<<(BS * NSEQ + 255)/256, 256, 0, stream>>>(o1, Wq2, Wk2, Wv2, q2, k2, v2);
    attn2_kernel<<<BS * (NSEQ/16),         32, 0, stream>>>(q2, k2, v2, mask, attn2, weight2, out);
}
